// LinearEncoder_4363686773057
// MI455X (gfx1250) — matrix operand order repaired and checked by reading, not compiled, not run on hardware
//
#include <hip/hip_runtime.h>
#include <hip/hip_bf16.h>

// ---------------------------------------------------------------------------
// GCNConv for MI455X (gfx1250): deg^-1/2 norm, bf16-split WMMA GEMM (fp32
// emulation: x@W ~= xhi@Whi + xhi@Wlo + xlo@Whi, f32 accum), edge-parallel
// atomic scatter (h and out are L2-resident: 51MB each vs 192MB L2).
//
// W is pre-swizzled into the exact WMMA B-fragment lane layout so the GEMM
// loads each 16-element bf16 fragment as one 32B vector load (2x b128).
// ---------------------------------------------------------------------------

typedef __attribute__((ext_vector_type(16))) __bf16 v16bf;
typedef __attribute__((ext_vector_type(8)))  float  v8f;

#define IN_CH  256
#define OUT_CH 128
#define KTILES (IN_CH / 32)   // 8
#define NTILES (OUT_CH / 16)  // 8

// ---- split W (f32) into bf16 hi/lo, swizzled to B-fragment layout ---------
// Fragment order: [ktile][ntile][lane][elem], 16 bf16 (32B) per lane.
// B 32x16 bf16 layout (ISA 7.12.2): lane<16 -> N=lane, K=k0+0..7 & k0+16..23;
//                                   lane>=16 -> N=lane-16, K=k0+8..15 & k0+24..31.
__global__ void gcn_wsplit_kernel(const float* __restrict__ W,
                                  __bf16* __restrict__ Whi,
                                  __bf16* __restrict__ Wlo) {
    int i = blockIdx.x * blockDim.x + threadIdx.x;
    if (i < IN_CH * OUT_CH) {
        int k = i >> 7;       // row of W  [0,256)
        int n = i & 127;      // col of W  [0,128)
        float f  = W[i];
        __bf16 h = (__bf16)f;
        __bf16 l = (__bf16)(f - (float)h);
        int ktile = k >> 5;
        int ntile = n >> 4;
        int lane  = (n & 15) | (((k >> 3) & 1) << 4);
        int elem  = (k & 7) | (((k >> 4) & 1) << 3);
        size_t idx = ((((size_t)ktile * NTILES + ntile) * 32) + lane) * 16 + elem;
        Whi[idx] = h;
        Wlo[idx] = l;
    }
}

// ---- degree: init to 1.0 (self loop), count edges, rsqrt ------------------
__global__ void gcn_deg_init_kernel(float* __restrict__ deg, int n) {
    int i = blockIdx.x * blockDim.x + threadIdx.x;
    if (i < n) deg[i] = 1.0f;
}

__global__ void gcn_deg_count_kernel(const int* __restrict__ dst,
                                     float* __restrict__ deg, int n_edges) {
    int e = blockIdx.x * blockDim.x + threadIdx.x;
    if (e < n_edges) atomicAdd(&deg[dst[e]], 1.0f);
}

__global__ void gcn_dinv_kernel(const float* __restrict__ deg,
                                float* __restrict__ dinv, int n) {
    int i = blockIdx.x * blockDim.x + threadIdx.x;
    if (i < n) dinv[i] = rsqrtf(deg[i]);
}

// ---- WMMA GEMM: h_scaled[m, :] = dinv[m] * (x[m, :] @ W) ------------------
// Block = 256 threads = 8 waves. Wave w owns rows [blk*128 + w*16, +16) and
// ALL 128 output columns (8 accumulators) so x is streamed exactly once.
__global__ void __launch_bounds__(256)
gcn_gemm_kernel(const float* __restrict__ x,
                const __bf16* __restrict__ Whi,
                const __bf16* __restrict__ Wlo,
                const float* __restrict__ dinv,
                float* __restrict__ h_scaled,
                float* __restrict__ out_acc,
                int n_nodes) {
    const int lane = threadIdx.x & 31;
    const int wave = threadIdx.x >> 5;
    const int half = lane >> 4;   // 0: lanes 0-15, 1: lanes 16-31
    const int l16  = lane & 15;

    const int m0 = blockIdx.x * 128 + wave * 16;

    // A-fragment row for this lane (clamped, never diverge around WMMA)
    int am = m0 + l16;
    if (am >= n_nodes) am = n_nodes - 1;
    const float* __restrict__ xrow = x + (size_t)am * IN_CH;

    // per-lane base of this wave's B fragments (32B per lane per fragment)
    const __bf16* __restrict__ whi_l = Whi + (size_t)lane * 16;
    const __bf16* __restrict__ wlo_l = Wlo + (size_t)lane * 16;

    v8f acc[NTILES];
#pragma unroll
    for (int t = 0; t < NTILES; ++t) acc[t] = (v8f){};

    for (int k0 = 0; k0 < IN_CH; k0 += 32) {
        // ---- A fragment (16x32 bf16): lane<16 holds K=k0+0..7 & k0+16..23,
        //      lane>=16 holds K=k0+8..15 & k0+24..31 (ISA 7.12.2 layout)
        const int ka = k0 + half * 8;
        float fa[16];
        {
            const float4* p0 = (const float4*)(xrow + ka);        // 32B aligned
            const float4* p1 = (const float4*)(xrow + ka + 16);
            float4 c0 = p0[0], c1 = p0[1], c2 = p1[0], c3 = p1[1];
            fa[0] = c0.x;  fa[1] = c0.y;  fa[2]  = c0.z;  fa[3]  = c0.w;
            fa[4] = c1.x;  fa[5] = c1.y;  fa[6]  = c1.z;  fa[7]  = c1.w;
            fa[8] = c2.x;  fa[9] = c2.y;  fa[10] = c2.z;  fa[11] = c2.w;
            fa[12] = c3.x; fa[13] = c3.y; fa[14] = c3.z;  fa[15] = c3.w;
        }
        v16bf a_hi, a_lo;
#pragma unroll
        for (int i = 0; i < 16; ++i) {
            __bf16 h = (__bf16)fa[i];
            a_hi[i] = h;
            a_lo[i] = (__bf16)(fa[i] - (float)h);
        }

        // ---- B fragments: one 32B vector load each (pre-swizzled layout)
        const size_t kbase = (size_t)(k0 >> 5) * NTILES;
#pragma unroll
        for (int t = 0; t < NTILES; ++t) {
            const size_t off = (kbase + t) * 32 * 16;   // bf16 elements
            v16bf b_hi = *(const v16bf*)(whi_l + off);
            v16bf b_lo = *(const v16bf*)(wlo_l + off);
            // 3-term fp32 emulation, f32 accumulate
            acc[t] = __builtin_amdgcn_wmma_f32_16x16x32_bf16(
                false, a_hi, false, b_hi, (short)0, acc[t], false, false);
            acc[t] = __builtin_amdgcn_wmma_f32_16x16x32_bf16(
                false, a_hi, false, b_lo, (short)0, acc[t], false, false);
            acc[t] = __builtin_amdgcn_wmma_f32_16x16x32_bf16(
                false, a_lo, false, b_hi, (short)0, acc[t], false, false);
        }
    }

    // ---- epilogue: C/D layout: lane -> N=l16, VGPR v -> M = half*8 + v
#pragma unroll
    for (int v = 0; v < 8; ++v) {
        int m = m0 + half * 8 + v;
        if (m < n_nodes) {
            float s = dinv[m];
#pragma unroll
            for (int t = 0; t < NTILES; ++t) {
                float val = acc[t][v] * s;
                size_t idx = (size_t)m * OUT_CH + t * 16 + l16;
                h_scaled[idx] = val;
                out_acc[idx]  = val;   // self-loop term pre-seeded
            }
        }
    }
}

// ---- edge scatter: one wave32 per edge, float4 per lane, 4 f32 atomics ----
__global__ void __launch_bounds__(256)
gcn_scatter_kernel(const int* __restrict__ src,
                   const int* __restrict__ dst,
                   const float* __restrict__ h_scaled,
                   float* __restrict__ out_acc, int n_edges) {
    int e = blockIdx.x * (blockDim.x >> 5) + (threadIdx.x >> 5);
    if (e >= n_edges) return;
    int lane = threadIdx.x & 31;
    int s = src[e];            // wave-uniform -> scalar load
    int d = dst[e];
    const float4* __restrict__ hp =
        (const float4*)(h_scaled + (size_t)s * OUT_CH);
    float4 v = hp[lane];
    float* op = out_acc + (size_t)d * OUT_CH + lane * 4;
    atomicAdd(op + 0, v.x);    // fire-and-forget -> global_atomic_add_f32
    atomicAdd(op + 1, v.y);
    atomicAdd(op + 2, v.z);
    atomicAdd(op + 3, v.w);
}

// ---- finalize: out = dinv[i] * out + b ------------------------------------
__global__ void gcn_finalize_kernel(float* __restrict__ out,
                                    const float* __restrict__ dinv,
                                    const float* __restrict__ b,
                                    int n_nodes) {
    size_t i = (size_t)blockIdx.x * blockDim.x + threadIdx.x;
    size_t total = (size_t)n_nodes * OUT_CH;
    if (i < total) {
        int node = (int)(i >> 7);      // /128
        int c    = (int)(i & 127);
        out[i] = out[i] * dinv[node] + b[c];
    }
}

// ---------------------------------------------------------------------------
extern "C" void kernel_launch(void* const* d_in, const int* in_sizes, int n_in,
                              void* d_out, int out_size, void* d_ws, size_t ws_size,
                              hipStream_t stream) {
    const float* x   = (const float*)d_in[0];
    const int*   ei  = (const int*)d_in[1];     // [2, E]
    const float* W   = (const float*)d_in[2];   // [256, 128]
    const float* b   = (const float*)d_in[3];   // [128]
    float* out = (float*)d_out;

    const int n_nodes = in_sizes[0] / IN_CH;
    const int n_edges = in_sizes[1] / 2;
    const int* src = ei;            // edge_index[0] = source j
    const int* dst = ei + n_edges;  // edge_index[1] = target i

    // workspace carve-up (256B aligned)
    char* p = (char*)d_ws;
    auto carve = [&](size_t bytes) {
        char* q = p;
        p += (bytes + 255) & ~(size_t)255;
        return q;
    };
    float*  deg  = (float*)carve((size_t)n_nodes * 4);
    float*  dinv = (float*)carve((size_t)n_nodes * 4);
    float*  hsc  = (float*)carve((size_t)n_nodes * OUT_CH * 4);
    __bf16* Whi  = (__bf16*)carve((size_t)IN_CH * OUT_CH * 2);
    __bf16* Wlo  = (__bf16*)carve((size_t)IN_CH * OUT_CH * 2);
    (void)ws_size; (void)n_in; (void)out_size;

    const int wn = IN_CH * OUT_CH;
    gcn_wsplit_kernel<<<(wn + 255) / 256, 256, 0, stream>>>(W, Whi, Wlo);

    gcn_deg_init_kernel<<<(n_nodes + 255) / 256, 256, 0, stream>>>(deg, n_nodes);
    gcn_deg_count_kernel<<<(n_edges + 255) / 256, 256, 0, stream>>>(dst, deg, n_edges);
    gcn_dinv_kernel<<<(n_nodes + 255) / 256, 256, 0, stream>>>(deg, dinv, n_nodes);

    // GEMM: 128 rows per block (8 waves x 16 rows), full 128-col output
    int gemm_blocks = (n_nodes + 127) / 128;
    gcn_gemm_kernel<<<gemm_blocks, 256, 0, stream>>>(x, Whi, Wlo, dinv,
                                                     hsc, out, n_nodes);

    // one wave per edge
    int edges_per_block = 256 / 32;
    int scat_blocks = (n_edges + edges_per_block - 1) / edges_per_block;
    gcn_scatter_kernel<<<scat_blocks, 256, 0, stream>>>(src, dst, hsc, out,
                                                        n_edges);

    size_t total = (size_t)n_nodes * OUT_CH;
    gcn_finalize_kernel<<<(int)((total + 255) / 256), 256, 0, stream>>>(
        out, dinv, b, n_nodes);
}
